// YunaAudioAttention_48619029791070
// MI455X (gfx1250) — matrix operand order repaired and checked by reading, not compiled, not run on hardware
//
#include <hip/hip_runtime.h>
#include <hip/hip_bf16.h>
#include <stdint.h>

typedef __attribute__((ext_vector_type(16))) _Float16 v16h;
typedef __attribute__((ext_vector_type(8)))  _Float16 v8h;
typedef __attribute__((ext_vector_type(8)))  float    v8f;

#define T_TOK 8192
#define E_DIM 1024
#define H_NUM 16
#define D_DIM 64
#define NSEG  8
#define L_SEG 1024

// ---------------------------------------------------------------------------
// CDNA5 async global->LDS copy (ASYNCcnt-tracked), 16B per lane.
// ---------------------------------------------------------------------------
__device__ __forceinline__ void async_copy_b128(unsigned lds_off, const void* g) {
    asm volatile("global_load_async_to_lds_b128 %0, %1, off"
                 :: "v"(lds_off), "v"((unsigned long long)g) : "memory");
}
__device__ __forceinline__ void wait_asynccnt0() {
    asm volatile("s_wait_asynccnt 0x0" ::: "memory");
}
__device__ __forceinline__ unsigned lds_addr_of(const void* p) {
    // generic pointers to LDS carry the AS3 offset in the low 32 bits
    return (unsigned)(uintptr_t)p;
}

// ---------------------------------------------------------------------------
// fp32 -> f16 conversion (elementwise)
// ---------------------------------------------------------------------------
__global__ void cvt_f32_to_f16_kernel(const float* __restrict__ in,
                                      _Float16* __restrict__ out, int n) {
    int i = blockIdx.x * blockDim.x + threadIdx.x;
    if (i < n) out[i] = (_Float16)in[i];
}

// ---------------------------------------------------------------------------
// fp32 weight (E x E) -> f16 * scale, transposed AND swizzled into K panels:
//   wsw[(k>>5)*(E*32) + j*32 + (k&31)] = w[j*E + k] * scale
// A 32(K) x 128(N) GEMM tile is one contiguous 8KB span, async-loadable
// straight into the [n][k32] LDS layout that B-fragments read as b128s.
// scale = D^-0.5 for wq (folds the attention scaling into the projection).
// ---------------------------------------------------------------------------
__global__ __launch_bounds__(256) void cvt_transpose_swz_w_kernel(
    const float* __restrict__ w, _Float16* __restrict__ wsw, float scale) {
    __shared__ float tile[32][33];
    const int bx = blockIdx.x * 32, by = blockIdx.y * 32;
    const int tx = threadIdx.x & 31, ty = threadIdx.x >> 5; // 32x8
#pragma unroll
    for (int i = 0; i < 32; i += 8)
        tile[ty + i][tx] = w[(size_t)(by + ty + i) * E_DIM + bx + tx];
    __syncthreads();
#pragma unroll
    for (int i = 0; i < 32; i += 8) {
        const int kk = bx + ty + i;   // transposed row (K index)
        const int jj = by + tx;       // transposed col (N index)
        wsw[(size_t)(kk >> 5) * (E_DIM * 32) + (size_t)jj * 32 + (kk & 31)] =
            (_Float16)(tile[tx][ty + i] * scale);
    }
}

// ---------------------------------------------------------------------------
// Async double-buffered f16 WMMA GEMM:  C(MxN) = A(MxK) * Bsw (+bias*bscale)
// A row-major f16; Bsw in K-panel swizzled layout (see above).
// 128x128 tile, 256 threads (8 waves); wave (wm,wn) computes 32x64:
//   2 M-tiles x 4 N-tiles -> 8 WMMAs per 32-K chunk, 12 ds_load_b128.
// Tiles staged with global_load_async_to_lds_b128, 2-deep pipeline.
// ---------------------------------------------------------------------------
template <bool HAS_BIAS, bool OUT16>
__global__ __launch_bounds__(256) void gemm_async_wmma_kernel(
    const _Float16* __restrict__ A, const _Float16* __restrict__ Bsw,
    const float* __restrict__ bias, float bscale, _Float16* __restrict__ C16,
    float* __restrict__ C32, int M, int N, int K) {
    __shared__ __align__(16) _Float16 As[2][128 * 32]; // [m][k] row-major
    __shared__ __align__(16) _Float16 Bs[2][128 * 32]; // [n][k] panel

    const int tid  = threadIdx.x;
    const int lane = tid & 31;
    const int wave = tid >> 5;
    const int hlf  = lane >> 4;
    const int l16  = lane & 15;
    const int wm   = wave & 3;   // 4 wave rows  * 32 = 128
    const int wn   = wave >> 2;  // 2 wave cols  * 64 = 128
    const int bm   = blockIdx.y * 128;
    const int bn   = blockIdx.x * 128;

    // per-thread async staging coordinates
    const int ar = tid >> 1;            // A tile row
    const int ac = (tid & 1) * 16;      // A tile col (halves)

    v8f zero = {};
    v8f acc[2][4];
#pragma unroll
    for (int mt = 0; mt < 2; ++mt)
#pragma unroll
        for (int nt = 0; nt < 4; ++nt) acc[mt][nt] = zero;

    const int nk = K >> 5;

    // ---- issue tile kt into buffer buf (4 x b128 per thread) ----
    auto issue_tile = [&](int kt, int buf) {
        const _Float16* ga = &A[(size_t)(bm + ar) * K + kt * 32 + ac];
        unsigned la = lds_addr_of(&As[buf][ar * 32 + ac]);
        async_copy_b128(la,      ga);
        async_copy_b128(la + 16, ga + 8);
        const _Float16* gb =
            &Bsw[(size_t)kt * ((size_t)N * 32) + (size_t)bn * 32 + tid * 16];
        unsigned lb = lds_addr_of(&Bs[buf][tid * 16]);
        async_copy_b128(lb,      gb);
        async_copy_b128(lb + 16, gb + 8);
    };

    issue_tile(0, 0);
    for (int kt = 0; kt < nk; ++kt) {
        const int buf = kt & 1;
        wait_asynccnt0();
        __syncthreads();
        if (kt + 1 < nk) issue_tile(kt + 1, buf ^ 1);

        union frag { v16h v; v8h h[2]; };
        frag af[2];
#pragma unroll
        for (int mt = 0; mt < 2; ++mt) {
            const int arow = wm * 32 + mt * 16 + l16;
            af[mt].h[0] = *(v8h*)&As[buf][arow * 32 + hlf * 8];
            af[mt].h[1] = *(v8h*)&As[buf][arow * 32 + 16 + hlf * 8];
        }
#pragma unroll
        for (int nt = 0; nt < 4; ++nt) {
            frag bf;
            const int bcol = wn * 64 + nt * 16 + l16;
            bf.h[0] = *(v8h*)&Bs[buf][bcol * 32 + hlf * 8];
            bf.h[1] = *(v8h*)&Bs[buf][bcol * 32 + 16 + hlf * 8];
#pragma unroll
            for (int mt = 0; mt < 2; ++mt)
                acc[mt][nt] = __builtin_amdgcn_wmma_f32_16x16x32_f16(
                    false, af[mt].v, false, bf.v, (short)0, acc[mt][nt],
                    false, false);
        }
    }

    // ---- straight-line epilogue (templated, no branches) ----
#pragma unroll
    for (int mt = 0; mt < 2; ++mt) {
        const int rbase = bm + wm * 32 + mt * 16 + hlf * 8;
#pragma unroll
        for (int nt = 0; nt < 4; ++nt) {
            const int col = bn + wn * 64 + nt * 16 + l16;
            float b = 0.0f;
            if constexpr (HAS_BIAS) b = bias[col] * bscale;
#pragma unroll
            for (int v = 0; v < 8; ++v) {
                const float val = acc[mt][nt][v] + b;
                const size_t off = (size_t)(rbase + v) * N + col;
                if constexpr (OUT16) C16[off] = (_Float16)val;
                else                 C32[off] = val;
            }
        }
    }
}

// ---------------------------------------------------------------------------
// Fused flash attention per (seg, head, 64-query block).
// 128 threads = 4 waves; each wave owns 16 query rows, full D=64.
// Q is pre-scaled by D^-0.5 (folded into wq/bq), so Q and K tiles are
// straight async global->LDS copies; V is transposed through VGPRs.
// ---------------------------------------------------------------------------
__global__ __launch_bounds__(128) void attn_wmma_kernel(
    const _Float16* __restrict__ Qg, const _Float16* __restrict__ Kg,
    const _Float16* __restrict__ Vg, const float* __restrict__ mask,
    _Float16* __restrict__ Og) {
    __shared__ __align__(16) _Float16 Qs[64 * 64];      // [q][d]
    __shared__ __align__(16) _Float16 Ks[64 * 64];      // [k][d] == B-frag layout
    __shared__ __align__(16) _Float16 Vs[64 * 64];      // [d][k] (transposed)
    __shared__ __align__(16) _Float16 Ps[4][16 * 64];   // per-wave P staging

    const int tid  = threadIdx.x;
    const int lane = tid & 31;
    const int wave = tid >> 5;
    const int hlf  = lane >> 4;
    const int l16  = lane & 15;

    const int seg  = blockIdx.z;
    const int head = blockIdx.y;
    const int q0   = blockIdx.x * 64;
    const size_t tokbase = (size_t)seg * L_SEG;
    const int fbase = head * D_DIM;

    const int sr = tid >> 1;             // staging row
    const int sc = (tid & 1) * 32;       // staging col (halves)

    // ---- async-stage Q tile (already scaled via wq) ----
    {
        const _Float16* src = &Qg[(tokbase + q0 + sr) * E_DIM + fbase + sc];
        unsigned lq = lds_addr_of(&Qs[sr * 64 + sc]);
#pragma unroll
        for (int i = 0; i < 4; ++i)
            async_copy_b128(lq + i * 16, src + i * 8);
    }

    float mrow[8], lrow[8];
    v8f zero = {};
    v8f oacc[4];
#pragma unroll
    for (int v = 0; v < 8; ++v) { mrow[v] = -3.0e38f; lrow[v] = 0.0f; }
#pragma unroll
    for (int nt = 0; nt < 4; ++nt) oacc[nt] = zero;

    for (int kb = 0; kb < L_SEG / 64; ++kb) {
        __syncthreads();
        // ---- async-stage K (row-major), VGPR-transpose V into [d][k] ----
        {
            const _Float16* ksrc =
                &Kg[(tokbase + kb * 64 + sr) * E_DIM + fbase + sc];
            unsigned lk = lds_addr_of(&Ks[sr * 64 + sc]);
#pragma unroll
            for (int i = 0; i < 4; ++i)
                async_copy_b128(lk + i * 16, ksrc + i * 8);

            int k = tid & 63, dseg = (tid >> 6) * 32;
            const _Float16* vsrc =
                &Vg[(tokbase + kb * 64 + k) * E_DIM + fbase + dseg];
#pragma unroll
            for (int i = 0; i < 32; ++i)
                Vs[(dseg + i) * 64 + k] = vsrc[i];
        }
        wait_asynccnt0();
        __syncthreads();

        // ---- S = Q @ K^T : 4 key-tiles x 2 k-chunks ----
        v8f sacc[4];
#pragma unroll
        for (int nt = 0; nt < 4; ++nt) sacc[nt] = zero;
#pragma unroll
        for (int kc = 0; kc < 2; ++kc) {
            union { v16h v; v8h h[2]; } af;
            const int arow = wave * 16 + l16;
            af.h[0] = *(v8h*)&Qs[arow * 64 + kc * 32 + hlf * 8];
            af.h[1] = *(v8h*)&Qs[arow * 64 + kc * 32 + 16 + hlf * 8];
#pragma unroll
            for (int nt = 0; nt < 4; ++nt) {
                union { v16h v; v8h h[2]; } bf;
                const int bcol = nt * 16 + l16;
                bf.h[0] = *(v8h*)&Ks[bcol * 64 + kc * 32 + hlf * 8];
                bf.h[1] = *(v8h*)&Ks[bcol * 64 + kc * 32 + 16 + hlf * 8];
                sacc[nt] = __builtin_amdgcn_wmma_f32_16x16x32_f16(
                    false, af.v, false, bf.v, (short)0, sacc[nt], false, false);
            }
        }

        // ---- add attention mask block ----
#pragma unroll
        for (int nt = 0; nt < 4; ++nt) {
#pragma unroll
            for (int v = 0; v < 8; ++v) {
                const int r = q0 + wave * 16 + v + 8 * hlf;
                const int c = kb * 64 + nt * 16 + l16;
                sacc[nt][v] += mask[(tokbase + r) * (size_t)T_TOK + tokbase + c];
            }
        }

        // ---- online softmax (row = v + 8*hlf lives across 16 lanes) ----
        float mnew[8], alpha[8], rsum[8];
#pragma unroll
        for (int v = 0; v < 8; ++v) {
            float m = fmaxf(fmaxf(sacc[0][v], sacc[1][v]),
                            fmaxf(sacc[2][v], sacc[3][v]));
#pragma unroll
            for (int off = 8; off >= 1; off >>= 1)
                m = fmaxf(m, __shfl_xor(m, off, 32));
            mnew[v]  = fmaxf(mrow[v], m);
            alpha[v] = __expf(mrow[v] - mnew[v]);
            mrow[v]  = mnew[v];
            rsum[v]  = 0.0f;
        }
        // ---- P = exp(S - m); stage to per-wave LDS in A-frag form ----
#pragma unroll
        for (int nt = 0; nt < 4; ++nt) {
#pragma unroll
            for (int v = 0; v < 8; ++v) {
                const float p = __expf(sacc[nt][v] - mnew[v]);
                rsum[v] += p;
                Ps[wave][(v + 8 * hlf) * 64 + nt * 16 + l16] = (_Float16)p;
            }
        }
#pragma unroll
        for (int v = 0; v < 8; ++v) {
            float s = rsum[v];
#pragma unroll
            for (int off = 8; off >= 1; off >>= 1)
                s += __shfl_xor(s, off, 32);
            lrow[v] = lrow[v] * alpha[v] + s;
#pragma unroll
            for (int nt = 0; nt < 4; ++nt) oacc[nt][v] *= alpha[v];
        }

        // ---- O += P @ V  (same-wave LDS RAW is in-order; no barrier) ----
#pragma unroll
        for (int kc = 0; kc < 2; ++kc) {
            union { v16h v; v8h h[2]; } af;
            af.h[0] = *(v8h*)&Ps[wave][l16 * 64 + kc * 32 + hlf * 8];
            af.h[1] = *(v8h*)&Ps[wave][l16 * 64 + kc * 32 + 16 + hlf * 8];
#pragma unroll
            for (int nt = 0; nt < 4; ++nt) {
                union { v16h v; v8h h[2]; } bf;
                const int bcol = nt * 16 + l16;
                bf.h[0] = *(v8h*)&Vs[bcol * 64 + kc * 32 + hlf * 8];
                bf.h[1] = *(v8h*)&Vs[bcol * 64 + kc * 32 + 16 + hlf * 8];
                oacc[nt] = __builtin_amdgcn_wmma_f32_16x16x32_f16(
                    false, af.v, false, bf.v, (short)0, oacc[nt], false, false);
            }
        }
    }

    // ---- normalize and write O as f16 ----
#pragma unroll
    for (int v = 0; v < 8; ++v) {
        const float inv = 1.0f / lrow[v];
        const int r = q0 + wave * 16 + v + 8 * hlf;
#pragma unroll
        for (int nt = 0; nt < 4; ++nt) {
            const int d = nt * 16 + l16;
            Og[(tokbase + r) * E_DIM + fbase + d] =
                (_Float16)(oacc[nt][v] * inv);
        }
    }
}

// ---------------------------------------------------------------------------
extern "C" void kernel_launch(void* const* d_in, const int* in_sizes, int n_in,
                              void* d_out, int out_size, void* d_ws, size_t ws_size,
                              hipStream_t stream) {
    (void)in_sizes; (void)n_in; (void)out_size; (void)ws_size;
    const float* hs   = (const float*)d_in[0];
    const float* mask = (const float*)d_in[2];
    const float* wq   = (const float*)d_in[3];
    const float* bq   = (const float*)d_in[4];
    const float* wk   = (const float*)d_in[5];
    const float* wv   = (const float*)d_in[6];
    const float* bv   = (const float*)d_in[7];
    const float* wo   = (const float*)d_in[8];
    const float* bo   = (const float*)d_in[9];
    float* out = (float*)d_out;

    // workspace layout (72 MB total); attn output reuses hs16 region
    char* ws = (char*)d_ws;
    _Float16* hs16 = (_Float16*)(ws);
    _Float16* wqS  = (_Float16*)(ws + (size_t)(16 << 20));
    _Float16* wkS  = (_Float16*)(ws + (size_t)(18 << 20));
    _Float16* wvS  = (_Float16*)(ws + (size_t)(20 << 20));
    _Float16* woS  = (_Float16*)(ws + (size_t)(22 << 20));
    _Float16* q16  = (_Float16*)(ws + (size_t)(24 << 20));
    _Float16* k16  = (_Float16*)(ws + (size_t)(40 << 20));
    _Float16* v16  = (_Float16*)(ws + (size_t)(56 << 20));
    _Float16* attn16 = hs16;

    const float qscale = 0.125f; // D^-0.5, folded into wq/bq

    const int nelem = T_TOK * E_DIM;
    cvt_f32_to_f16_kernel<<<nelem / 256, 256, 0, stream>>>(hs, hs16, nelem);

    dim3 tg(E_DIM / 32, E_DIM / 32);
    cvt_transpose_swz_w_kernel<<<tg, 256, 0, stream>>>(wq, wqS, qscale);
    cvt_transpose_swz_w_kernel<<<tg, 256, 0, stream>>>(wk, wkS, 1.0f);
    cvt_transpose_swz_w_kernel<<<tg, 256, 0, stream>>>(wv, wvS, 1.0f);
    cvt_transpose_swz_w_kernel<<<tg, 256, 0, stream>>>(wo, woS, 1.0f);

    dim3 gg(E_DIM / 128, T_TOK / 128);
    gemm_async_wmma_kernel<true,  true><<<gg, 256, 0, stream>>>(
        hs16, wqS, bq, qscale, q16, nullptr, T_TOK, E_DIM, E_DIM);
    gemm_async_wmma_kernel<false, true><<<gg, 256, 0, stream>>>(
        hs16, wkS, nullptr, 1.0f, k16, nullptr, T_TOK, E_DIM, E_DIM);
    gemm_async_wmma_kernel<true,  true><<<gg, 256, 0, stream>>>(
        hs16, wvS, bv, 1.0f, v16, nullptr, T_TOK, E_DIM, E_DIM);

    attn_wmma_kernel<<<dim3(L_SEG / 64, H_NUM, NSEG), 128, 0, stream>>>(
        q16, k16, v16, mask, attn16);

    gemm_async_wmma_kernel<true,  false><<<gg, 256, 0, stream>>>(
        attn16, woS, bo, 1.0f, nullptr, out, T_TOK, E_DIM, E_DIM);
}
